// nonConsecutiveConvLayer3_56796647522676
// MI455X (gfx1250) — matrix operand order repaired and checked by reading, not compiled, not run on hardware
//
#include <hip/hip_runtime.h>
#include <cmath>

// ---- problem constants (match reference) ----
#define B_  64
#define L_  256
#define D_  1024
#define F_  1024

typedef __attribute__((ext_vector_type(16))) __bf16 v16bf;
typedef __attribute__((ext_vector_type(8)))  __bf16 v8bf;
typedef __attribute__((ext_vector_type(8)))  float  v8f;
typedef __attribute__((ext_vector_type(4)))  float  v4f;

// Block = 4 waves; wave kw owns K slice [kw*256, kw*256+256).
// Double-buffered LDS: xt = x tile per timestep (bf16), pbuf = partial exchange.
__shared__ __bf16 g_xt[2][16 * D_];        // 2 x 32768 B
__shared__ float  g_pbuf[2][3 * 3 * 256];  // 2 x 9216 B  ([kw-1][mat][lane*8..])

__global__ __launch_bounds__(128, 1)
void maxplus_scan_wmma(const float* __restrict__ x,
                       const float* __restrict__ W0,
                       const float* __restrict__ W1,
                       const float* __restrict__ W2,
                       const float* __restrict__ bias,
                       float* __restrict__ out)
{
    const int tid  = threadIdx.x;
    const int lane = tid & 31;
    const int kw   = tid >> 5;     // wave id == K-slice id (0..3)
    const int half = lane >> 4;
    const int l15  = lane & 15;

    const int f0 = blockIdx.x * 16;   // F/16 = 64 blocks in x
    const int b0 = blockIdx.y * 16;   // B/16 = 4 blocks in y

    // ---------------------------------------------------------------
    // Stationary B operands (W tiles) from global into registers,
    // f32 -> bf16, WMMA B layout: lane (half,l15) holds column N = l15,
    // K rows half*8+0..7 and 16+half*8+0..7 of the k-step.
    // 3 mats x 8 ksteps x 8 VGPRs = 192 VGPRs resident for the whole scan.
    // ---------------------------------------------------------------
    v16bf barr[3][8];
    #pragma unroll
    for (int mat = 0; mat < 3; ++mat) {
        const float* wsrc = (mat == 0) ? W0 : (mat == 1) ? W1 : W2;
        #pragma unroll
        for (int ks = 0; ks < 8; ++ks) {
            const int krow = kw * 256 + ks * 32 + half * 8;
            const float* src = wsrc + (size_t)krow * F_ + f0 + l15;
            v16bf bb;
            #pragma unroll
            for (int s = 0; s < 8; ++s) {
                bb[s]     = (__bf16)src[(size_t)s * F_];
                bb[8 + s] = (__bf16)src[(size_t)(16 + s) * F_];
            }
            barr[mat][ks] = bb;
        }
    }

    // per-lane bias column + persistent hidden state (wave 0 owns the scan)
    const float bval = bias[f0 + l15];
    v8f h1 = {}, h2 = {}, h3 = {};

    const int d_base = kw * 256;

    // ---- stage x[:,l,:] tile: f32 -> bf16 into g_xt[pb] (all 128 threads) ----
    auto stage = [&](int l, int pb) {
        #pragma unroll 4
        for (int idx = tid; idx < (16 * D_) / 8; idx += 128) {
            const int m  = idx >> 7;
            const int d8 = (idx & 127) * 8;
            const float* src = x + ((size_t)(b0 + m) * L_ + l) * D_ + d8;
            v4f xa = *(const v4f*)(src);
            v4f xb = *(const v4f*)(src + 4);
            v8bf xc;
            xc[0] = (__bf16)xa[0]; xc[1] = (__bf16)xa[1];
            xc[2] = (__bf16)xa[2]; xc[3] = (__bf16)xa[3];
            xc[4] = (__bf16)xb[0]; xc[5] = (__bf16)xb[1];
            xc[6] = (__bf16)xb[2]; xc[7] = (__bf16)xb[3];
            *(v8bf*)(g_xt[pb] + m * D_ + d8) = xc;
        }
    };

    stage(0, 0);
    __syncthreads();

    for (int l = 0; l < L_; ++l) {
        const int p = l & 1;

        // Prefetch next timestep's x tile into the other buffer.
        if (l + 1 < L_) stage(l + 1, p ^ 1);

        // ---- software-pipelined WMMA burst: loads(ks+1) are fenced ahead of
        //      the WMMA triple of ks, so the DS wait before each triple is a
        //      partial wait already covered by the previous triple's latency ----
        const __bf16* xrow = g_xt[p] + l15 * D_ + d_base;
        auto loadA = [&](int ks) -> v16bf {
            v8bf lo = *(const v8bf*)(xrow + ks * 32 + half * 8);
            v8bf hi = *(const v8bf*)(xrow + ks * 32 + 16 + half * 8);
            return __builtin_shufflevector(lo, hi,
                       0,1,2,3,4,5,6,7,8,9,10,11,12,13,14,15);
        };

        v8f c0 = {}, c1 = {}, c2 = {};
        v16bf aA = loadA(0);
        v16bf aB;
        #pragma unroll
        for (int ks = 0; ks < 8; ++ks) {
            const bool even = (ks & 1) == 0;
            v16bf acur = even ? aA : aB;
            if (ks < 7) {                    // prefetch next A slice
                if (even) aB = loadA(ks + 1);
                else      aA = loadA(ks + 1);
            }
            __builtin_amdgcn_sched_barrier(0);  // keep prefetch above the WMMAs
            c0 = __builtin_amdgcn_wmma_f32_16x16x32_bf16(false, acur, false,
                     barr[0][ks], (short)0, c0, false, false);
            c1 = __builtin_amdgcn_wmma_f32_16x16x32_bf16(false, acur, false,
                     barr[1][ks], (short)0, c1, false, false);
            c2 = __builtin_amdgcn_wmma_f32_16x16x32_bf16(false, acur, false,
                     barr[2][ks], (short)0, c2, false, false);
            __builtin_amdgcn_sched_barrier(0);  // cap live A buffers at two
        }

        // ---- K-split: waves 1..3 publish partials into this parity's pbuf ----
        if (kw != 0) {
            float* pb = g_pbuf[p] + (kw - 1) * 3 * 256;
            *(v4f*)(pb + 0 * 256 + lane * 8)     = __builtin_shufflevector(c0, c0, 0,1,2,3);
            *(v4f*)(pb + 0 * 256 + lane * 8 + 4) = __builtin_shufflevector(c0, c0, 4,5,6,7);
            *(v4f*)(pb + 1 * 256 + lane * 8)     = __builtin_shufflevector(c1, c1, 0,1,2,3);
            *(v4f*)(pb + 1 * 256 + lane * 8 + 4) = __builtin_shufflevector(c1, c1, 4,5,6,7);
            *(v4f*)(pb + 2 * 256 + lane * 8)     = __builtin_shufflevector(c2, c2, 0,1,2,3);
            *(v4f*)(pb + 2 * 256 + lane * 8 + 4) = __builtin_shufflevector(c2, c2, 4,5,6,7);
        }

        __syncthreads();  // partials + next x tile visible

        // ---- wave 0: reduce K-split partials and advance the max-plus scan ----
        if (kw == 0) {
            float pr0[8] = {}, pr1[8] = {}, pr2[8] = {};
            #pragma unroll
            for (int w = 0; w < 3; ++w) {
                const float* pb = g_pbuf[p] + w * 3 * 256;
                v4f q;
                q = *(const v4f*)(pb + 0*256 + lane*8);     pr0[0]+=q[0]; pr0[1]+=q[1]; pr0[2]+=q[2]; pr0[3]+=q[3];
                q = *(const v4f*)(pb + 0*256 + lane*8 + 4); pr0[4]+=q[0]; pr0[5]+=q[1]; pr0[6]+=q[2]; pr0[7]+=q[3];
                q = *(const v4f*)(pb + 1*256 + lane*8);     pr1[0]+=q[0]; pr1[1]+=q[1]; pr1[2]+=q[2]; pr1[3]+=q[3];
                q = *(const v4f*)(pb + 1*256 + lane*8 + 4); pr1[4]+=q[0]; pr1[5]+=q[1]; pr1[6]+=q[2]; pr1[7]+=q[3];
                q = *(const v4f*)(pb + 2*256 + lane*8);     pr2[0]+=q[0]; pr2[1]+=q[1]; pr2[2]+=q[2]; pr2[3]+=q[3];
                q = *(const v4f*)(pb + 2*256 + lane*8 + 4); pr2[4]+=q[0]; pr2[5]+=q[1]; pr2[6]+=q[2]; pr2[7]+=q[3];
            }
            #pragma unroll
            for (int j = 0; j < 8; ++j) {
                const float a0 = c0[j] + pr0[j];
                const float a1 = c1[j] + pr1[j];
                const float a2 = c2[j] + pr2[j];
                const float m1 = fmaxf(a0, h1[j]);
                const float m2 = fmaxf(h1[j] + a1, h2[j]);
                const float m3 = fmaxf(h2[j] + a2, h3[j]);
                h1[j] = m1; h2[j] = m2; h3[j] = m3;
            }
        }
    }

    // ---- epilogue: tanh(h3 + b); C layout: lane n = l15, row m = j + half*8 ----
    if (kw == 0) {
        #pragma unroll
        for (int j = 0; j < 8; ++j) {
            const int m = j + half * 8;
            out[(size_t)(b0 + m) * F_ + f0 + l15] = tanhf(h3[j] + bval);
        }
    }
}

extern "C" void kernel_launch(void* const* d_in, const int* in_sizes, int n_in,
                              void* d_out, int out_size, void* d_ws, size_t ws_size,
                              hipStream_t stream) {
    (void)in_sizes; (void)n_in; (void)d_ws; (void)ws_size; (void)out_size;
    const float* x  = (const float*)d_in[0];
    const float* W0 = (const float*)d_in[1];
    const float* W1 = (const float*)d_in[2];
    const float* W2 = (const float*)d_in[3];
    const float* b  = (const float*)d_in[4];
    float* out = (float*)d_out;

    dim3 grid(F_ / 16, B_ / 16);  // (64, 4) blocks; 4 waves each; 1024 waves
    dim3 block(128);
    maxplus_scan_wmma<<<grid, block, 0, stream>>>(x, W0, W1, W2, b, out);
}
